// MultiHeadDifAttention_3736621547672
// MI455X (gfx1250) — matrix operand order repaired and checked by reading, not compiled, not run on hardware
//
#include <hip/hip_runtime.h>
#include <hip/hip_bf16.h>
#include <math.h>

// ---------------------------------------------------------------------------
// MI455X (gfx1250, wave32) differential multi-head attention.
//
//   1. f32->f16 converts of x, Wq, Wk, Wv, Wo
//   2. qbuf = x@Wq^T, kbuf = x@Wk^T, vbuf = x@Wv^T     (WMMA GEMM, f16 out)
//   3. vT   = per-(b,h) transpose of v                  (so attn@V is A@B^T)
//   4. fused diff-attention, 4 waves/WG sharing K via TDM-staged LDS chunks
//      (tensor_load_to_lds + s_wait_tensorcnt, double buffered), per-lane
//      online softmax (no cross-lane ops in the hot loop), attn -> d_out,
//      attn@V via LDS fragment re-layout -> y (f32 ws)
//   5. group-norm stats per (b, s-block-of-256)  [matches reference reshape]
//   6. gather + normalize + (1-lambda_init) -> act (f16)
//   7. out = act@Wo^T + bo -> d_out (f32)
// ---------------------------------------------------------------------------

typedef _Float16 v8h  __attribute__((ext_vector_type(8)));
typedef _Float16 v16h __attribute__((ext_vector_type(16)));
typedef float    v8f  __attribute__((ext_vector_type(8)));
typedef unsigned int u32x4 __attribute__((ext_vector_type(4)));
typedef unsigned int u32x8 __attribute__((ext_vector_type(8)));

union HV { v16h v; v8h h[2]; };

// A-fragment (16xK, 16-bit): lane l<16 -> row l, halves [0..7]=K+0..7,
// [8..15]=K+16..23 ; lane>=16 -> row l-16, K+8..15 / K+24..31.
__device__ __forceinline__ v16h ld_fragA(const _Float16* __restrict__ p,
                                         int ld, int mb, int k0) {
  int l = threadIdx.x & 31;
  const _Float16* q = p + (size_t)(mb + (l & 15)) * ld + k0 + ((l >> 4) << 3);
  HV u;
  u.h[0] = *(const v8h*)q;
  u.h[1] = *(const v8h*)(q + 16);
  return u.v;
}

// B-fragment for C=A@B^T with B row-major [N x K]: lane l -> column nb+(l&15)
// of B^T = row of B; halves K+0..7 / K+16..23 (lane>=16: K+8..15 / K+24..31).
__device__ __forceinline__ v16h ld_fragB(const _Float16* __restrict__ p,
                                         int ld, int nb, int k0) {
  int l = threadIdx.x & 31;
  const _Float16* q = p + (size_t)(nb + (l & 15)) * ld + k0 + ((l >> 4) << 3);
  HV u;
  u.h[0] = *(const v8h*)q;
  u.h[1] = *(const v8h*)(q + 16);
  return u.v;
}

__device__ __forceinline__ v8f wmma16(v16h a, v16h b, v8f c) {
  return __builtin_amdgcn_wmma_f32_16x16x32_f16(false, a, false, b,
                                                (short)0, c, false, false);
}

// Tensor Data Mover: 1-D copy of one 16KB K-chunk (32 rows x 512B, rows
// contiguous) from global into LDS.  D# per CDNA5 ISA ch8 (gfx1250):
//   group0: count=1 | lds_addr | global_addr[56:0] | type=2
//   group1: data_size=3 (8B units), tensor_dim0=2048, tensor_dim1=1,
//           tile_dim0=2048, tile_dim1/2=0, tensor_dim0_stride=2048
__device__ __forceinline__ void tdm_load_k(const _Float16* gsrc,
                                           _Float16* ldst) {
  unsigned long long ga = (unsigned long long)gsrc;
  unsigned lds = (unsigned)(unsigned long long)ldst;  // flat lo32 == LDS off
  u32x4 g0 = {1u, lds, (unsigned)ga,
              ((unsigned)(ga >> 32) & 0x01FFFFFFu) | 0x80000000u};
  u32x8 g1 = {(3u << 16),      // data_size = 8B units
              (2048u << 16),   // tensor_dim0[15:0] in [31:16]
              (1u << 16),      // tensor_dim0 hi=0 | tensor_dim1[15:0]=1
              (2048u << 16),   // tensor_dim1 hi=0 | tile_dim0 = 2048
              0u,              // tile_dim1 = 0, tile_dim2 = 0
              2048u,           // tensor_dim0_stride lo
              0u, 0u};
  u32x4 gz = {0u, 0u, 0u, 0u};
  asm volatile("tensor_load_to_lds %0, %1, %2, %3"
               :: "s"(g0), "s"(g1), "s"(gz), "s"(gz)
               : "memory");
}

// ------------------------- elementwise helpers -----------------------------

__global__ void f32_to_f16_kernel(const float* __restrict__ in,
                                  _Float16* __restrict__ out, int n) {
  int i = blockIdx.x * blockDim.x + threadIdx.x;
  if (i < n) out[i] = (_Float16)in[i];
}

// vT[((b*8+h)*128 + d)*2048 + s] = v[b,h,s,d]
__global__ void transpose_v_kernel(const _Float16* __restrict__ vbuf,
                                   _Float16* __restrict__ vT) {
  int idx = blockIdx.x * blockDim.x + threadIdx.x;  // < 2*8*128*2048
  int s  = idx & 2047;
  int t  = idx >> 11;
  int d  = t & 127;
  int bh = t >> 7;
  int h  = bh & 7;
  int b  = bh >> 3;
  vT[idx] = vbuf[(size_t)b * (2048 * 1024) + (size_t)h * (2048 * 128) +
                 (size_t)s * 128 + d];
}

// ------------------------- generic WMMA GEMM -------------------------------
// C[M x N] = A[M x K] @ B[N x K]^T + bias[col]; one wave per 16x64 tile.
template <typename OutT>
__global__ __launch_bounds__(32) void gemm_abT_kernel(
    const _Float16* __restrict__ A, const _Float16* __restrict__ B,
    const float* __restrict__ bias, OutT* __restrict__ C,
    int K, int lda, int ldb, int ldc) {
  int mb = blockIdx.x * 16;
  int nb = blockIdx.y * 64;
  v8f acc[4] = {};
  for (int k0 = 0; k0 < K; k0 += 32) {
    if (k0 + 32 < K)
      __builtin_prefetch(A + (size_t)(mb + (threadIdx.x & 15)) * lda + k0 + 32,
                         0, 3);
    v16h a = ld_fragA(A, lda, mb, k0);
#pragma unroll
    for (int j = 0; j < 4; ++j) {
      v16h bf = ld_fragB(B, ldb, nb + j * 16, k0);
      acc[j] = wmma16(a, bf, acc[j]);
    }
  }
  int l = threadIdx.x & 31;
#pragma unroll
  for (int j = 0; j < 4; ++j)
#pragma unroll
    for (int r = 0; r < 8; ++r) {
      int row = mb + r + ((l >> 4) << 3);
      int col = nb + j * 16 + (l & 15);
      float v = acc[j][r] + bias[col];
      C[(size_t)row * ldc + col] = (OutT)v;
    }
}

// ------------------------- fused diff attention ----------------------------
// 4 waves / workgroup, 64 queries per WG for one (b,h).  K streamed in 32-key
// chunks (both score channels) through double-buffered LDS by the TDM.
__global__ __launch_bounds__(128) void diffattn_kernel(
    const _Float16* __restrict__ qbuf, const _Float16* __restrict__ kbuf,
    const _Float16* __restrict__ vT, const float* __restrict__ lam_p,
    float* __restrict__ attn_out, float* __restrict__ y_out) {
  const int S = 2048, Dh = 128, H = 8;
  const int NC = S / 32;  // 64 chunks of 32 keys
  const float scale = 0.08838834764831845f;  // 1/sqrt(128)
  int wv = threadIdx.x >> 5;
  int l = threadIdx.x & 31;
  int h = blockIdx.y, b = blockIdx.z;
  int qb = blockIdx.x * 64 + wv * 16;
  const _Float16* qph = qbuf + (size_t)b * (S * 2048) + (size_t)h * (S * 256);
  const _Float16* kph = kbuf + (size_t)b * (S * 2048) + (size_t)h * (S * 256);
  const _Float16* vph = vT + (size_t)(b * H + h) * Dh * S;  // [128 x S]
  const float lam = lam_p[0];
  const bool w0 = (wv == 0);

  __shared__ __attribute__((aligned(16))) _Float16 ktile[2][32 * 256];  // 32KB
  __shared__ __attribute__((aligned(16))) _Float16 atile[4][16 * 32];   //  4KB

  v16h q1a[4], q2a[4];
#pragma unroll
  for (int t = 0; t < 4; ++t) {
    q1a[t] = ld_fragA(qph, 256, qb, t * 32);
    q2a[t] = ld_fragA(qph + 128, 256, qb, t * 32);
  }

  float m1[8], s1[8], m2[8], s2[8];
#pragma unroll
  for (int r = 0; r < 8; ++r) {
    m1[r] = -1e30f; s1[r] = 0.f;
    m2[r] = -1e30f; s2[r] = 0.f;
  }

  // ---- pass A: per-lane online (max,sum); no cross-lane ops in the loop ---
  if (w0) tdm_load_k(kph, &ktile[0][0]);
  for (int kc = 0; kc < NC; ++kc) {
    int cur = kc & 1;
    __syncthreads();  // readers of ktile[1-cur] from previous iter are done
    if (w0) {
      if (kc + 1 < NC) {
        tdm_load_k(kph + (size_t)(kc + 1) * 32 * 256, &ktile[1 - cur][0]);
        __builtin_amdgcn_s_wait_tensorcnt(1);  // chunk kc landed
      } else {
        __builtin_amdgcn_s_wait_tensorcnt(0);
      }
    }
    __syncthreads();  // ktile[cur] visible to all waves
    const _Float16* kt = &ktile[cur][0];
#pragma unroll
    for (int sub = 0; sub < 2; ++sub) {
      v8f c1 = {}, c2 = {};
#pragma unroll
      for (int t = 0; t < 4; ++t) {
        c1 = wmma16(q1a[t], ld_fragB(kt, 256, sub * 16, t * 32), c1);
        c2 = wmma16(q2a[t], ld_fragB(kt, 256, sub * 16, 128 + t * 32), c2);
      }
#pragma unroll
      for (int r = 0; r < 8; ++r) {
        float x1 = c1[r] * scale;
        float n1 = fmaxf(m1[r], x1);
        s1[r] = s1[r] * __expf(m1[r] - n1) + __expf(x1 - n1);
        m1[r] = n1;
        float x2 = c2[r] * scale;
        float n2 = fmaxf(m2[r], x2);
        s2[r] = s2[r] * __expf(m2[r] - n2) + __expf(x2 - n2);
        m2[r] = n2;
      }
    }
  }

  // ---- merge (m,s) across the 16-lane row group; invert sums once --------
#pragma unroll
  for (int r = 0; r < 8; ++r) {
    float m = m1[r], s = s1[r];
    for (int o = 8; o >= 1; o >>= 1) {
      float mo = __shfl_xor(m, o, 16);
      float so = __shfl_xor(s, o, 16);
      float nm = fmaxf(m, mo);
      s = s * __expf(m - nm) + so * __expf(mo - nm);
      m = nm;
    }
    m1[r] = m; s1[r] = 1.f / s;
    m = m2[r]; s = s2[r];
    for (int o = 8; o >= 1; o >>= 1) {
      float mo = __shfl_xor(m, o, 16);
      float so = __shfl_xor(s, o, 16);
      float nm = fmaxf(m, mo);
      s = s * __expf(m - nm) + so * __expf(mo - nm);
      m = nm;
    }
    m2[r] = m; s2[r] = 1.f / s;
  }

  // ---- pass B: attn output + attn@V (same TDM pipeline) -------------------
  v8f oacc[8] = {};
  if (w0) tdm_load_k(kph, &ktile[0][0]);
  for (int kc = 0; kc < NC; ++kc) {
    int cur = kc & 1;
    __syncthreads();
    if (w0) {
      if (kc + 1 < NC) {
        tdm_load_k(kph + (size_t)(kc + 1) * 32 * 256, &ktile[1 - cur][0]);
        __builtin_amdgcn_s_wait_tensorcnt(1);
      } else {
        __builtin_amdgcn_s_wait_tensorcnt(0);
      }
    }
    __syncthreads();
    const _Float16* kt = &ktile[cur][0];
#pragma unroll
    for (int sub = 0; sub < 2; ++sub) {
      int kb = kc * 2 + sub;
      v8f c1 = {}, c2 = {};
#pragma unroll
      for (int t = 0; t < 4; ++t) {
        c1 = wmma16(q1a[t], ld_fragB(kt, 256, sub * 16, t * 32), c1);
        c2 = wmma16(q2a[t], ld_fragB(kt, 256, sub * 16, 128 + t * 32), c2);
      }
#pragma unroll
      for (int r = 0; r < 8; ++r) {
        float p1 = __expf(c1[r] * scale - m1[r]) * s1[r];
        float p2 = __expf(c2[r] * scale - m2[r]) * s2[r];
        float a = p1 - lam * p2;
        int row = r + ((l >> 4) << 3);
        int col = kb * 16 + (l & 15);
        attn_out[((size_t)(b * H + h) * S + qb + row) * S + col] = a;
        atile[wv][row * 32 + sub * 16 + (l & 15)] = (_Float16)a;
      }
    }
    // per-wave private atile: DS ops are in-order within a wave; make sure
    // the b16 stores are complete before the b128 fragment reads.
    asm volatile("s_wait_dscnt 0x0" ::: "memory");
    {
      const _Float16* ap = &atile[wv][0] + (l & 15) * 32 + ((l >> 4) << 3);
      HV u;
      u.h[0] = *(const v8h*)ap;
      u.h[1] = *(const v8h*)(ap + 16);
      v16h aa = u.v;
#pragma unroll
      for (int n = 0; n < 8; ++n) {
        v16h vb = ld_fragB(vph, S, n * 16, kc * 32);
        oacc[n] = wmma16(aa, vb, oacc[n]);
      }
    }
  }

#pragma unroll
  for (int n = 0; n < 8; ++n)
#pragma unroll
    for (int r = 0; r < 8; ++r) {
      int row = qb + r + ((l >> 4) << 3);
      int col = n * 16 + (l & 15);
      y_out[((size_t)(b * H + h) * S + row) * Dh + col] = oacc[n][r];
    }
}

// ------------------- group-norm (matches reference reshape) ----------------
__global__ void gn_stats_kernel(const float* __restrict__ y,
                                float* __restrict__ stats) {
  int gi = blockIdx.x;  // [0,16)
  int b = gi >> 3, g = gi & 7;
  const int per = 8 * 256 * 128;  // 262144
  float sum = 0.f, sq = 0.f;
  int tid = threadIdx.x;
  for (int i = tid; i < per; i += 256) {
    int d = i & 127;
    int t = i >> 7;
    int s = g * 256 + (t & 255);
    int h2 = t >> 8;
    float v = y[((size_t)(b * 8 + h2) * 2048 + s) * 128 + d];
    sum += v; sq += v * v;
  }
  __shared__ float rs[256], rq[256];
  rs[tid] = sum; rq[tid] = sq;
  __syncthreads();
  for (int st = 128; st > 0; st >>= 1) {
    if (tid < st) { rs[tid] += rs[tid + st]; rq[tid] += rq[tid + st]; }
    __syncthreads();
  }
  if (tid == 0) {
    float mean = rs[0] / per;
    float var = rq[0] / per - mean * mean;
    stats[gi * 2] = mean;
    stats[gi * 2 + 1] = rsqrtf(var + 1e-5f);
  }
}

__global__ void gn_apply_kernel(const float* __restrict__ y,
                                const float* __restrict__ stats,
                                const float* __restrict__ gnw,
                                const float* __restrict__ gnb,
                                const float* __restrict__ lam_init,
                                _Float16* __restrict__ act, int total) {
  int idx = blockIdx.x * blockDim.x + threadIdx.x;
  if (idx >= total) return;
  const int S = 2048;
  int b = idx / (S * 1024);
  int rem = idx - b * (S * 1024);
  int s2 = rem >> 10;
  int c = rem & 1023;
  int h2 = c >> 7;
  int d = c & 127;
  int n = h2 * S + s2;
  int s = n >> 3;
  int hh = n & 7;
  int g = s >> 8;
  float mean = stats[(b * 8 + g) * 2];
  float rstd = stats[(b * 8 + g) * 2 + 1];
  float v = y[((size_t)(b * 8 + hh) * S + s) * 128 + d];
  v = (v - mean) * rstd * gnw[g] + gnb[g];
  v *= (1.f - lam_init[0]);
  act[idx] = (_Float16)v;
}

// ---------------------------------------------------------------------------

extern "C" void kernel_launch(void* const* d_in, const int* in_sizes, int n_in,
                              void* d_out, int out_size, void* d_ws,
                              size_t ws_size, hipStream_t stream) {
  (void)in_sizes; (void)n_in; (void)out_size; (void)ws_size;
  const int B = 2, S = 2048, H = 8, Dh = 128;
  const int M = B * S;  // 4096

  const float* x   = (const float*)d_in[0];
  const float* Wq  = (const float*)d_in[1];
  const float* bq  = (const float*)d_in[2];
  const float* Wk  = (const float*)d_in[3];
  const float* bk  = (const float*)d_in[4];
  const float* Wv  = (const float*)d_in[5];
  const float* bv  = (const float*)d_in[6];
  const float* Wo  = (const float*)d_in[7];
  const float* bo  = (const float*)d_in[8];
  const float* lam = (const float*)d_in[9];
  const float* lin = (const float*)d_in[10];
  const float* gnw = (const float*)d_in[11];
  const float* gnb = (const float*)d_in[12];

  float* outp = (float*)d_out;                    // (B,S,1024) : 4194304
  float* attn = outp + (size_t)B * S * (H * Dh);  // (B,H,S,S)  : 67108864

  char* ws = (char*)d_ws;
  _Float16* x16  = (_Float16*)(ws + 0);
  _Float16* wq16 = (_Float16*)(ws + 1048576);
  _Float16* wk16 = (_Float16*)(ws + 1572864);
  _Float16* wv16 = (_Float16*)(ws + 2097152);
  _Float16* wo16 = (_Float16*)(ws + 2359296);
  _Float16* qbuf = (_Float16*)(ws + 2621440);
  _Float16* kbuf = (_Float16*)(ws + 19398656);
  _Float16* vbuf = (_Float16*)(ws + 36175872);
  _Float16* vT   = (_Float16*)(ws + 44564480);
  float*    yws  = (float*)   (ws + 52953088);
  _Float16* act  = (_Float16*)(ws + 69730304);
  float*    st   = (float*)   (ws + 78118912);

  auto cvt = [&](const float* in, _Float16* out, int n) {
    f32_to_f16_kernel<<<(n + 255) / 256, 256, 0, stream>>>(in, out, n);
  };
  cvt(x, x16, M * Dh);
  cvt(Wq, wq16, 2048 * 128);
  cvt(Wk, wk16, 2048 * 128);
  cvt(Wv, wv16, 1024 * 128);
  cvt(Wo, wo16, 128 * 1024);

  gemm_abT_kernel<_Float16><<<dim3(M / 16, 2048 / 64), 32, 0, stream>>>(
      x16, wq16, bq, qbuf, 128, 128, 128, 2048);
  gemm_abT_kernel<_Float16><<<dim3(M / 16, 2048 / 64), 32, 0, stream>>>(
      x16, wk16, bk, kbuf, 128, 128, 128, 2048);
  gemm_abT_kernel<_Float16><<<dim3(M / 16, 1024 / 64), 32, 0, stream>>>(
      x16, wv16, bv, vbuf, 128, 128, 128, 1024);

  {
    int n = B * H * Dh * S;
    transpose_v_kernel<<<(n + 255) / 256, 256, 0, stream>>>(vbuf, vT);
  }

  diffattn_kernel<<<dim3(S / 64, H, B), 128, 0, stream>>>(qbuf, kbuf, vT, lam,
                                                          attn, yws);

  gn_stats_kernel<<<16, 256, 0, stream>>>(yws, st);

  {
    int n = B * S * H * Dh;
    gn_apply_kernel<<<(n + 255) / 256, 256, 0, stream>>>(yws, st, gnw, gnb,
                                                         lin, act, n);
  }

  gemm_abT_kernel<float><<<dim3(M / 16, 128 / 64), 32, 0, stream>>>(
      act, wo16, bo, outp, 1024, 1024, 1024, 128);
}
